// ETNN_1898375545098
// MI455X (gfx1250) — compile-verified, matrix operand order checked by reading, not run on hardware
//
#include <hip/hip_runtime.h>

// ---------------------------------------------------------------------------
// Types for CDNA5 WMMA (wave32): A/B = 16 halves/lane (16x32 / 32x16 tiles),
// C/D = 8 floats/lane (16x16 tile).
// ---------------------------------------------------------------------------
typedef __attribute__((ext_vector_type(16))) _Float16 v16h;
typedef __attribute__((ext_vector_type(8)))  _Float16 v8h;
typedef __attribute__((ext_vector_type(8)))  float    v8f;
typedef __attribute__((ext_vector_type(4)))  int      v4i;

#define BGRAPH 512
#define HDIM   256

// gfx1250 async global->LDS path (ASYNCcnt-tracked DMA), guarded so a missing
// builtin falls back to the known-good synchronous load+ds_store copy.
#if defined(__gfx1250__) && __has_builtin(__builtin_amdgcn_global_load_async_to_lds_b128)
#define USE_ASYNC_LDS 1
#else
#define USE_ASYNC_LDS 0
#endif

// Builtin signature (from hipcc diagnostic): param0 is v4i in the global
// (__device__ / addrspace(1)) AS; param1 is the LDS-side v4i pointer.
typedef __attribute__((address_space(1))) v4i* gv4i_p;
typedef __attribute__((address_space(3))) v4i* lv4i_p;

__device__ __forceinline__ void wait_async_zero() {
#if USE_ASYNC_LDS
#if __has_builtin(__builtin_amdgcn_s_wait_asynccnt)
  __builtin_amdgcn_s_wait_asynccnt(0);
#else
  asm volatile("s_wait_asynccnt 0x0" ::: "memory");
#endif
#endif
}

// Order-preserving float <-> uint encoding for atomicMax on floats.
__device__ __forceinline__ unsigned enc_f32(float f) {
  unsigned u = __float_as_uint(f);
  return (u & 0x80000000u) ? ~u : (u | 0x80000000u);
}
__device__ __forceinline__ float dec_f32(unsigned u) {
  unsigned bits = (u & 0x80000000u) ? (u ^ 0x80000000u) : ~u;
  return __uint_as_float(bits);
}
__device__ __forceinline__ float silu_f(float x) { return x / (1.0f + __expf(-x)); }

// ---------------------------------------------------------------------------
// Pack a row-major f32 [K, Ncol] matrix into f16 WMMA-B fragments:
// layout [kt][nt][lane 0..31][elem 0..15], 512 halves (1 KB) per 32x16 tile.
// Lane l owns column nt*16+(l&15); elems 0..7 -> K = kt*32 + (l>>4)*8 + e,
// elems 8..15 -> K = same + 16.
// ---------------------------------------------------------------------------
__global__ __launch_bounds__(256) void pack_b_kernel(
    const float* __restrict__ src, _Float16* __restrict__ dst, int K, int Ncol) {
  int tid = blockIdx.x * 256 + threadIdx.x;
  int total = K * Ncol;
  if (tid >= total) return;
  int blockI = tid >> 9;         // /512  : tile id
  int within = tid & 511;
  int lane = within >> 4;
  int e    = within & 15;
  int NT = Ncol >> 4;
  int kt = blockI / NT, nt = blockI % NT;
  int k   = kt * 32 + ((lane >> 4) * 8) + (e & 7) + ((e >= 8) ? 16 : 0);
  int col = nt * 16 + (lane & 15);
  dst[tid] = (_Float16)src[(size_t)k * Ncol + col];
}

// Stage one 32x16-column-tile group (8 kt fragments, 8 KB) into LDS.
// Each of the 256 threads moves 32 bytes (two B128 transfers).
__device__ __forceinline__ void stage_tile(const _Float16* __restrict__ w1p,
                                           int nt, _Float16* dst, int tid) {
  const int kt = tid >> 5, bl = tid & 31;
  const _Float16* src = w1p + ((size_t)(kt * 16 + nt) * 512) + (size_t)bl * 16;
  _Float16* d = dst + ((size_t)(kt * 32 + bl) * 16);
#if USE_ASYNC_LDS
  __builtin_amdgcn_global_load_async_to_lds_b128((gv4i_p)src, (lv4i_p)d, 0, 0);
  __builtin_amdgcn_global_load_async_to_lds_b128((gv4i_p)src, (lv4i_p)d, 16, 0);
#else
  *(v16h*)d = *(const v16h*)src;
#endif
}

// ---------------------------------------------------------------------------
// Score kernel: s[n] = tanh(h @ W1 + b1) @ w2 + b2   (TAU == 1)
// One wave per 16-row tile, 8 waves/block. A fragments (K=256) hoisted in
// registers (f32->f16); W1 fragments double-buffer-staged into LDS via async
// DMA; tanh/dot epilogue fused on the in-register accumulator.
// ---------------------------------------------------------------------------
__global__ __launch_bounds__(256) void score_kernel(
    const float* __restrict__ h, int N,
    const _Float16* __restrict__ w1p,   // packed [8 kt][16 nt][32][16]
    const float* __restrict__ b1,       // [256]
    const float* __restrict__ w2,       // [256]
    const float* __restrict__ b2p,      // [1]
    float* __restrict__ s_out) {
  __shared__ _Float16 lds_b[2][8 * 512] __attribute__((aligned(32)));  // 16 KB

  const int tid  = threadIdx.x;
  const int lane = tid & 31;
  const int wave = tid >> 5;
  const int m0   = (blockIdx.x * 8 + wave) * 16;

  const int row_ = m0 + (lane & 15);
  const int row  = (row_ < N) ? row_ : (N - 1);   // clamp: keep EXEC full
  const int koff = (lane >> 4) * 8;

  // Kick off the first W1 stage before touching h.
  stage_tile(w1p, 0, lds_b[0], tid);

  // Hoist the 8 A fragments covering K = 0..255 for this 16-row tile.
  v16h a[8];
  const float* hrow = h + (size_t)row * HDIM;
  #pragma unroll
  for (int kt = 0; kt < 8; ++kt) {
    const float* p = hrow + kt * 32 + koff;
    #pragma unroll
    for (int i = 0; i < 8; ++i) a[kt][i] = (_Float16)p[i];
    #pragma unroll
    for (int i = 0; i < 8; ++i) a[kt][8 + i] = (_Float16)p[16 + i];
  }

  v8f sacc = {};                 // per-lane partial of s over owned entries
  for (int nt = 0; nt < 16; ++nt) {
    _Float16* cur = lds_b[nt & 1];
    wait_async_zero();           // my staged transfers for this buffer done
    __syncthreads();             // everyone's transfers visible
    if (nt + 1 < 16) stage_tile(w1p, nt + 1, lds_b[(nt + 1) & 1], tid);

    v8f acc = {};
    #pragma unroll
    for (int kt = 0; kt < 8; ++kt) {
      v16h bfrag = *(const v16h*)(&cur[(kt * 32 + lane) * 16]);
      acc = __builtin_amdgcn_wmma_f32_16x16x32_f16(
          false, a[kt], false, bfrag, (short)0, acc, false, false);
    }
    // Epilogue: +b1, tanh, * w2[col], accumulate.  Lane owns col nt*16+(l&15).
    const int   col = nt * 16 + (lane & 15);
    const float bb  = b1[col];
    const float ww  = w2[col];
    #pragma unroll
    for (int v = 0; v < 8; ++v) sacc[v] += tanhf(acc[v] + bb) * ww;
  }

  // Reduce over the 16 columns each half-wave owns.
  #pragma unroll
  for (int m = 1; m < 16; m <<= 1) {
    #pragma unroll
    for (int v = 0; v < 8; ++v) sacc[v] += __shfl_xor(sacc[v], m, 32);
  }
  if ((lane & 15) == 0) {
    const float b2 = b2p[0];
    const int rbase = m0 + (lane >> 4) * 8;   // C layout: low half rows 0..7
    #pragma unroll
    for (int v = 0; v < 8; ++v) {
      int r = rbase + v;
      if (r < N) s_out[r] = sacc[v] + b2;
    }
  }
}

// ---------------------------------------------------------------------------
// Segment-softmax stats.
// ---------------------------------------------------------------------------
__global__ __launch_bounds__(256) void smax_kernel(
    const float* __restrict__ s, const int* __restrict__ b, int N,
    unsigned* __restrict__ smaxEnc) {
  int i = blockIdx.x * 256 + threadIdx.x;
  if (i >= N) return;
  atomicMax(&smaxEnc[b[i]], enc_f32(s[i]));
}

__global__ __launch_bounds__(256) void denom_kernel(
    const float* __restrict__ s, const int* __restrict__ b, int N,
    const unsigned* __restrict__ smaxEnc, float* __restrict__ denom) {
  int i = blockIdx.x * 256 + threadIdx.x;
  if (i >= N) return;
  int g = b[i];
  atomicAdd(&denom[g], __expf(s[i] - dec_f32(smaxEnc[g])));
}

// ---------------------------------------------------------------------------
// Aggregation: ssum / att / max / cnt per graph. b is sorted, so a block
// accumulates a run of same-graph cells in registers and flushes atomics only
// at run boundaries. One thread per feature column (fully coalesced h reads).
// ---------------------------------------------------------------------------
__global__ __launch_bounds__(256) void agg_kernel(
    const float* __restrict__ h, const int* __restrict__ b, int N,
    const float* __restrict__ s, const unsigned* __restrict__ smaxEnc,
    const float* __restrict__ denom,
    float* __restrict__ ssum, float* __restrict__ att,
    unsigned* __restrict__ mxEnc, float* __restrict__ cnt, int chunk) {
  const int j  = threadIdx.x;
  int c0 = blockIdx.x * chunk;
  if (c0 >= N) return;
  int c1 = c0 + chunk; if (c1 > N) c1 = N;

  int   curg = -1;
  float aS = 0.f, aA = 0.f, aM = -3.0e38f, aC = 0.f;
  for (int c = c0; c < c1; ++c) {
    int g = b[c];
    if (g != curg) {
      if (curg >= 0) {
        atomicAdd(&ssum[(size_t)curg * HDIM + j], aS);
        atomicAdd(&att [(size_t)curg * HDIM + j], aA);
        atomicMax(&mxEnc[(size_t)curg * HDIM + j], enc_f32(aM));
        if (j == 0) atomicAdd(&cnt[curg], aC);
      }
      curg = g; aS = aA = 0.f; aM = -3.0e38f; aC = 0.f;
    }
    __builtin_prefetch(h + (size_t)(c + 4) * HDIM + j, 0, 1);   // global_prefetch_b8
    float val = h[(size_t)c * HDIM + j];
    float w   = __expf(s[c] - dec_f32(smaxEnc[g])) / denom[g];
    aS += val; aA += w * val; aM = fmaxf(aM, val); aC += 1.0f;
  }
  atomicAdd(&ssum[(size_t)curg * HDIM + j], aS);
  atomicAdd(&att [(size_t)curg * HDIM + j], aA);
  atomicMax(&mxEnc[(size_t)curg * HDIM + j], enc_f32(aM));
  if (j == 0) atomicAdd(&cnt[curg], aC);
}

// ---------------------------------------------------------------------------
// Build the dense f16 agg matrix [B, 4H] per rank: [ssum | mean | max | att].
// ---------------------------------------------------------------------------
__global__ __launch_bounds__(256) void finalize_agg_kernel(
    const float* __restrict__ ssum, const float* __restrict__ att,
    const unsigned* __restrict__ mxEnc, const float* __restrict__ cnt,
    _Float16* __restrict__ agg16) {   // [3][512][1024]
  int g = blockIdx.x, r = blockIdx.y, j = threadIdx.x;
  size_t gi = (size_t)r * BGRAPH + g;
  float S = ssum[gi * HDIM + j];
  float C = cnt[gi];
  float mean = S / fmaxf(C, 1.0f);
  float M = (C > 0.0f) ? dec_f32(mxEnc[gi * HDIM + j]) : 0.0f;
  float A = att[gi * HDIM + j];
  _Float16* dst = agg16 + gi * (4 * HDIM);
  dst[j]            = (_Float16)S;
  dst[HDIM + j]     = (_Float16)mean;
  dst[2 * HDIM + j] = (_Float16)M;
  dst[3 * HDIM + j] = (_Float16)A;
}

// ---------------------------------------------------------------------------
// Projection: state[:, r*256:(r+1)*256] = agg[B,1024] @ pw[1024,256] + pb.
// WMMA f16, one wave per 16x16 output tile, K=1024 in 32 steps.
// ---------------------------------------------------------------------------
__global__ __launch_bounds__(256) void proj_kernel(
    const _Float16* __restrict__ agg16,   // [3][512][1024] row-major f16
    const _Float16* __restrict__ pwp,     // packed [3][32 kt][16 nt][32][16]
    const float* __restrict__ pb,         // [3][256]
    float* __restrict__ state) {          // [512][768]
  const int r    = blockIdx.y;
  const int tid  = threadIdx.x;
  const int lane = tid & 31;
  const int wave = tid >> 5;
  const int tileIdx = blockIdx.x * 8 + wave;  // 0..511
  const int mt = tileIdx >> 4;                // row tile 0..31
  const int nt = tileIdx & 15;                // col tile 0..15

  const _Float16* Abase = agg16 + (size_t)r * BGRAPH * 1024;
  const _Float16* Bbase = pwp + (size_t)r * 32 * 16 * 512;
  const int row   = mt * 16 + (lane & 15);
  const int koffL = (lane >> 4) * 8;

  v8f acc = {};
  for (int kt = 0; kt < 32; ++kt) {
    v16h a;
    const _Float16* ap = Abase + (size_t)row * 1024 + kt * 32 + koffL;
    v8h lo = *(const v8h*)ap;
    v8h hi = *(const v8h*)(ap + 16);
    #pragma unroll
    for (int i = 0; i < 8; ++i) { a[i] = lo[i]; a[8 + i] = hi[i]; }
    v16h bfrag =
        *(const v16h*)(Bbase + ((size_t)(kt * 16 + nt) * 512) + (size_t)lane * 16);
    acc = __builtin_amdgcn_wmma_f32_16x16x32_f16(
        false, a, false, bfrag, (short)0, acc, false, false);
  }
  const int col  = nt * 16 + (lane & 15);
  const float bias = pb[r * HDIM + col];
  const int rbase = mt * 16 + (lane >> 4) * 8;
  #pragma unroll
  for (int v = 0; v < 8; ++v)
    state[(size_t)(rbase + v) * 768 + r * HDIM + col] = acc[v] + bias;
}

// ---------------------------------------------------------------------------
// Final MLP: LayerNorm(768) -> silu -> f1[768,256] -> silu -> f2[256] -> out.
// One block per graph.
// ---------------------------------------------------------------------------
__global__ __launch_bounds__(256) void final_kernel(
    const float* __restrict__ state, const float* __restrict__ lng,
    const float* __restrict__ lnb, const float* __restrict__ f1w,
    const float* __restrict__ f1b, const float* __restrict__ f2w,
    const float* __restrict__ f2b, float* __restrict__ out) {
  __shared__ float xs[768];
  __shared__ float red[18];
  const int g = blockIdx.x, tid = threadIdx.x, lane = tid & 31, wv = tid >> 5;
  const float* row = state + (size_t)g * 768;
  float v0 = row[tid], v1 = row[tid + 256], v2 = row[tid + 512];
  float lsum = v0 + v1 + v2;
  float lsq  = v0 * v0 + v1 * v1 + v2 * v2;
  #pragma unroll
  for (int m = 1; m < 32; m <<= 1) {
    lsum += __shfl_xor(lsum, m, 32);
    lsq  += __shfl_xor(lsq, m, 32);
  }
  if (lane == 0) { red[wv] = lsum; red[8 + wv] = lsq; }
  __syncthreads();
  if (tid == 0) {
    float s = 0.f, q = 0.f;
    for (int i = 0; i < 8; ++i) { s += red[i]; q += red[8 + i]; }
    float mu = s / 768.0f;
    red[16] = mu;
    red[17] = q / 768.0f - mu * mu;
  }
  __syncthreads();
  const float mu = red[16], inv = rsqrtf(red[17] + 1e-5f);
  xs[tid]       = silu_f((v0 - mu) * inv * lng[tid]       + lnb[tid]);
  xs[tid + 256] = silu_f((v1 - mu) * inv * lng[tid + 256] + lnb[tid + 256]);
  xs[tid + 512] = silu_f((v2 - mu) * inv * lng[tid + 512] + lnb[tid + 512]);
  __syncthreads();
  float acc = f1b[tid];
  #pragma unroll 4
  for (int k = 0; k < 768; ++k) acc = fmaf(xs[k], f1w[k * 256 + tid], acc);
  float prod = silu_f(acc) * f2w[tid];
  #pragma unroll
  for (int m = 1; m < 32; m <<= 1) prod += __shfl_xor(prod, m, 32);
  if (lane == 0) red[wv] = prod;
  __syncthreads();
  if (tid == 0) {
    float s = 0.f;
    for (int i = 0; i < 8; ++i) s += red[i];
    out[g] = s + f2b[0];
  }
}

// ---------------------------------------------------------------------------
// Host launcher.
// ---------------------------------------------------------------------------
extern "C" void kernel_launch(void* const* d_in, const int* in_sizes, int n_in,
                              void* d_out, int out_size, void* d_ws, size_t ws_size,
                              hipStream_t stream) {
  (void)n_in; (void)out_size; (void)ws_size;
  const float* h[3] = {(const float*)d_in[0], (const float*)d_in[2],
                       (const float*)d_in[4]};
  const int* b[3] = {(const int*)d_in[1], (const int*)d_in[3],
                     (const int*)d_in[5]};
  int N[3] = {in_sizes[1], in_sizes[3], in_sizes[5]};
  const float* s1w = (const float*)d_in[6];
  const float* s1b = (const float*)d_in[7];
  const float* s2w = (const float*)d_in[8];
  const float* s2b = (const float*)d_in[9];
  const float* pw  = (const float*)d_in[10];
  const float* pb  = (const float*)d_in[11];
  const float* lng = (const float*)d_in[12];
  const float* lnb = (const float*)d_in[13];
  const float* f1w = (const float*)d_in[14];
  const float* f1b = (const float*)d_in[15];
  const float* f2w = (const float*)d_in[16];
  const float* f2b = (const float*)d_in[17];
  float* out = (float*)d_out;

  // Workspace layout (256-byte aligned slots).
  char* base = (char*)d_ws;
  size_t off = 0;
  auto take = [&](size_t bytes) -> size_t {
    size_t o = off;
    off = (off + bytes + 255) & ~(size_t)255;
    return o;
  };
  size_t o_smax  = take(3 * BGRAPH * sizeof(unsigned));
  size_t o_denom = take(3 * BGRAPH * sizeof(float));
  size_t o_cnt   = take(3 * BGRAPH * sizeof(float));
  size_t o_ssum  = take((size_t)3 * BGRAPH * HDIM * sizeof(float));
  size_t o_att   = take((size_t)3 * BGRAPH * HDIM * sizeof(float));
  size_t o_mx    = take((size_t)3 * BGRAPH * HDIM * sizeof(unsigned));
  size_t zeroEnd = off;
  size_t o_s[3];
  for (int r = 0; r < 3; ++r) o_s[r] = take((size_t)N[r] * sizeof(float));
  size_t o_w1p = take((size_t)3 * 256 * 256 * sizeof(_Float16));
  size_t o_pwp = take((size_t)3 * 1024 * 256 * sizeof(_Float16));
  size_t o_agg = take((size_t)3 * BGRAPH * 1024 * sizeof(_Float16));
  size_t o_st  = take((size_t)BGRAPH * 768 * sizeof(float));

  unsigned* smaxEnc = (unsigned*)(base + o_smax);
  float*    denom   = (float*)(base + o_denom);
  float*    cnt     = (float*)(base + o_cnt);
  float*    ssum    = (float*)(base + o_ssum);
  float*    att     = (float*)(base + o_att);
  unsigned* mxEnc   = (unsigned*)(base + o_mx);
  _Float16* w1p     = (_Float16*)(base + o_w1p);
  _Float16* pwp     = (_Float16*)(base + o_pwp);
  _Float16* agg16   = (_Float16*)(base + o_agg);
  float*    state   = (float*)(base + o_st);

  // Zero accumulators (encoded-max init 0u == most-negative float).
  (void)hipMemsetAsync(base, 0, zeroEnd, stream);

  // Pack W1 / proj_w to f16 WMMA-B fragment layout.
  for (int r = 0; r < 3; ++r) {
    int tw1 = 256 * 256;
    pack_b_kernel<<<(tw1 + 255) / 256, 256, 0, stream>>>(
        s1w + (size_t)r * tw1, w1p + (size_t)r * tw1, 256, 256);
    int tpw = 1024 * 256;
    pack_b_kernel<<<(tpw + 255) / 256, 256, 0, stream>>>(
        pw + (size_t)r * tpw, pwp + (size_t)r * tpw, 1024, 256);
  }

  // Per-rank scoring + softmax stats + aggregation.
  const int CH = 256;
  for (int r = 0; r < 3; ++r) {
    float* s_r = (float*)(base + o_s[r]);
    int tiles  = (N[r] + 15) / 16;
    score_kernel<<<(tiles + 7) / 8, 256, 0, stream>>>(
        h[r], N[r], w1p + (size_t)r * 256 * 256, s1b + r * HDIM, s2w + r * HDIM,
        s2b + r, s_r);
    int tg = (N[r] + 255) / 256;
    smax_kernel<<<tg, 256, 0, stream>>>(s_r, b[r], N[r], smaxEnc + r * BGRAPH);
    denom_kernel<<<tg, 256, 0, stream>>>(s_r, b[r], N[r],
                                         smaxEnc + r * BGRAPH, denom + r * BGRAPH);
    agg_kernel<<<(N[r] + CH - 1) / CH, 256, 0, stream>>>(
        h[r], b[r], N[r], s_r, smaxEnc + r * BGRAPH, denom + r * BGRAPH,
        ssum + (size_t)r * BGRAPH * HDIM, att + (size_t)r * BGRAPH * HDIM,
        mxEnc + (size_t)r * BGRAPH * HDIM, cnt + r * BGRAPH, CH);
  }

  finalize_agg_kernel<<<dim3(BGRAPH, 3), 256, 0, stream>>>(ssum, att, mxEnc, cnt,
                                                           agg16);
  proj_kernel<<<dim3(64, 3), 256, 0, stream>>>(agg16, pwp, pb, state);
  final_kernel<<<BGRAPH, 256, 0, stream>>>(state, lng, lnb, f1w, f1b, f2w, f2b,
                                           out);
}
